// GatedPropagationModule_22222160789650
// MI455X (gfx1250) — compile-verified, matrix operand order checked by reading, not run on hardware
//
#include <hip/hip_runtime.h>
#include <hip/hip_bf16.h>
#include <math.h>

// ---------------------------------------------------------------------------
// Problem constants (from reference): h=w=48, L=2304, bs=2
//   d_qk=256, d_vu=1024, NH=8, D_ATT=32, HID=128, C=1024, half=512
// Token-major matrices: MROWS = L*bs = 4608, row index r = l*2 + b
// ---------------------------------------------------------------------------
#define LQ      2304
#define BS      2
#define MROWS   4608      // L*bs
#define DQK     256
#define CDIM    1024
#define HALF    512
#define NH      8
#define HID     128
#define HW      48
#define INV_TSC 0.17677669529663687f   // 1/sqrt(32)

// Padded LDS row strides (avoid bank aliasing: 2304 % 64 == 0)
#define SROW    2312      // floats, scores strip
#define AROW    2312      // halves, AV attn strip

typedef __attribute__((ext_vector_type(16))) _Float16 v16h;
typedef __attribute__((ext_vector_type(8)))  float    v8f;

// ---------------------------------------------------------------------------
// WMMA fragment loaders (CDNA5 16x16x32 f16 layout, CDNA5 ISA §7.12.2)
// A-frag: 16x32 tile, row-major source with leading dim `ld` (elements).
//   lane 0-15 : row=lane,    K in {0..7,16..23}
//   lane 16-31: row=lane-16, K in {8..15,24..31}
// B-frag uses identical pattern from an [N][K]-major buffer (lane -> column N).
// Works for both global and LDS source pointers.
// ---------------------------------------------------------------------------
__device__ __forceinline__ v16h frag_ld_f16(const _Float16* __restrict__ base, int ld) {
  const int lane = threadIdx.x & 31;
  const int row  = lane & 15;
  const int kofs = (lane & 16) ? 8 : 0;
  const _Float16* p = base + (size_t)row * ld + kofs;
  union { unsigned int u[8]; v16h v; } r;
#pragma unroll
  for (int i = 0; i < 8; ++i) {
    const int k = 2 * i + ((i & 4) ? 8 : 0);
    r.u[i] = *reinterpret_cast<const unsigned int*>(p + k);  // 2 packed halves
  }
  return r.v;
}

// Same A-fragment but sourced from an f32 row-major matrix, converted on load.
__device__ __forceinline__ v16h frag_ld_f32(const float* __restrict__ base, int ld) {
  const int lane = threadIdx.x & 31;
  const int row  = lane & 15;
  const int kofs = (lane & 16) ? 8 : 0;
  const float* p = base + (size_t)row * ld + kofs;
  union { unsigned int u[8]; v16h v; } r;
#pragma unroll
  for (int i = 0; i < 8; ++i) {
    const int k = 2 * i + ((i & 4) ? 8 : 0);
    union { _Float16 h[2]; unsigned int u; } cv;
    cv.h[0] = (_Float16)p[k];
    cv.h[1] = (_Float16)p[k + 1];
    r.u[i] = cv.u;
  }
  return r.v;
}

__device__ __forceinline__ v8f wmma32(v16h a, v16h b, v8f c) {
  return __builtin_amdgcn_wmma_f32_16x16x32_f16(false, a, false, b, (short)0, c,
                                                false, false);
}

// C/D layout helpers
__device__ __forceinline__ int cd_row0() { return ((threadIdx.x & 31) & 16) ? 8 : 0; }
__device__ __forceinline__ int cd_col()  { return (threadIdx.x & 31) & 15; }

__device__ __forceinline__ float silu(float x) { return x / (1.0f + __expf(-x)); }

// ---------------------------------------------------------------------------
// Prep kernels: f32 -> f16 copy, and f32 [K][N] -> f16 [N][K] transpose
// ---------------------------------------------------------------------------
__global__ void k_cvt(const float* __restrict__ in, _Float16* __restrict__ out, int n) {
  int t = blockIdx.x * 256 + threadIdx.x;
  if (t < n) out[t] = (_Float16)in[t];
}

__global__ void k_cvt_tr(const float* __restrict__ in, _Float16* __restrict__ out,
                         int K, int N) {
  int t = blockIdx.x * 256 + threadIdx.x;
  if (t >= K * N) return;
  int k = t / N, n = t - k * N;
  out[(size_t)n * K + k] = (_Float16)in[t];
}

// ---------------------------------------------------------------------------
// GEMM 1: qkh = f16(Qh[4608,256] @ WqkT + b_qk)   (Wqk pre-transposed [N][K])
// ---------------------------------------------------------------------------
__global__ void k_gemm_qk(const _Float16* __restrict__ A, const _Float16* __restrict__ Bt,
                          const float* __restrict__ bias, _Float16* __restrict__ out) {
  const int wave = (blockIdx.x << 3) + (threadIdx.x >> 5);
  const int tilesN = DQK / 16;                       // 16
  if (wave >= (MROWS / 16) * tilesN) return;
  const int tm = wave / tilesN, tn = wave - tm * tilesN;
  const _Float16* Ap = A  + (size_t)(tm * 16) * DQK;
  const _Float16* Bp = Bt + (size_t)(tn * 16) * DQK;
  v8f acc = {};
#pragma unroll
  for (int k = 0; k < DQK; k += 32)
    acc = wmma32(frag_ld_f16(Ap + k, DQK), frag_ld_f16(Bp + k, DQK), acc);
  const int n = tn * 16 + cd_col();
  const int r0 = tm * 16 + cd_row0();
  const float bn = bias[n];
#pragma unroll
  for (int i = 0; i < 8; ++i)
    out[(size_t)(r0 + i) * DQK + n] = (_Float16)(acc[i] + bn);
}

// ---------------------------------------------------------------------------
// Gated branch GEMMs: X @ W + b, per-head interleave + silu.
//   native col n in [0,512): head hh=n>>6, j=(n&63)+jofs
//   V variant: writes f16 VgT[b][hh][j][l]  (transposed for AV B-operand)
//   U variant: writes f32 Ug[r][hh*128+j]
// A = (Vh|Uh) + colOfs, lda = 512. Bt = [512][256] f16.
// ---------------------------------------------------------------------------
__global__ void k_gemm_gated_v(const _Float16* __restrict__ A, const _Float16* __restrict__ Bt,
                               const float* __restrict__ bias, _Float16* __restrict__ VgT,
                               int jofs) {
  const int wave = (blockIdx.x << 3) + (threadIdx.x >> 5);
  if (wave >= (MROWS / 16) * (HALF / 16)) return;
  const int tm = wave / 32, tn = wave - tm * 32;
  const _Float16* Ap = A  + (size_t)(tm * 16) * HALF;
  const _Float16* Bp = Bt + (size_t)(tn * 16) * 256;
  v8f acc = {};
#pragma unroll
  for (int k = 0; k < 256; k += 32)
    acc = wmma32(frag_ld_f16(Ap + k, HALF), frag_ld_f16(Bp + k, 256), acc);
  const int n  = tn * 16 + cd_col();
  const int hh = n >> 6;
  const int j  = (n & 63) + jofs;
  const float bn = bias[n];
  const int r0 = tm * 16 + cd_row0();
#pragma unroll
  for (int i = 0; i < 8; ++i) {
    const int r = r0 + i, l = r >> 1, b = r & 1;
    VgT[(size_t)((b * NH + hh) * HID + j) * LQ + l] = (_Float16)silu(acc[i] + bn);
  }
}

__global__ void k_gemm_gated_u(const _Float16* __restrict__ A, const _Float16* __restrict__ Bt,
                               const float* __restrict__ bias, float* __restrict__ Ug,
                               int jofs) {
  const int wave = (blockIdx.x << 3) + (threadIdx.x >> 5);
  if (wave >= (MROWS / 16) * (HALF / 16)) return;
  const int tm = wave / 32, tn = wave - tm * 32;
  const _Float16* Ap = A  + (size_t)(tm * 16) * HALF;
  const _Float16* Bp = Bt + (size_t)(tn * 16) * 256;
  v8f acc = {};
#pragma unroll
  for (int k = 0; k < 256; k += 32)
    acc = wmma32(frag_ld_f16(Ap + k, HALF), frag_ld_f16(Bp + k, 256), acc);
  const int n  = tn * 16 + cd_col();
  const int hh = n >> 6;
  const int c  = hh * HID + (n & 63) + jofs;
  const float bn = bias[n];
  const int r0 = tm * 16 + cd_row0();
#pragma unroll
  for (int i = 0; i < 8; ++i)
    Ug[(size_t)(r0 + i) * CDIM + c] = silu(acc[i] + bn);
}

// ---------------------------------------------------------------------------
// FUSED scores + softmax.
// One block per (b,h, 16-row strip): 8 waves compute 144 16x16 score tiles
// (single K=32 WMMA each) into a 16 x SROW f32 strip held in LDS (uses the
// CDNA5 320KB-WGP LDS), then the block does the row softmax in LDS and writes
// the normalized attn strip to global exactly once.
// Dynamic LDS layout: S[16*SROW] f32 | Ah[16*32] f16 | red[256] f32 | rowv[16]
// ---------------------------------------------------------------------------
__global__ void k_scores_softmax(const _Float16* __restrict__ qkh,
                                 float* __restrict__ attn) {
  extern __shared__ char smem_raw[];
  float*    S    = (float*)smem_raw;
  _Float16* Ah   = (_Float16*)(smem_raw + (size_t)16 * SROW * 4);
  float*    red  = (float*)(Ah + 16 * 32);
  float*    rowv = red + 256;

  const int bh = blockIdx.x / (LQ / 16);
  const int tm = blockIdx.x - bh * (LQ / 16);
  const int b = bh >> 3, hh = bh & 7;
  const int cofs = b * DQK + hh * 32;
  const int t = threadIdx.x;

  // Stage shared A tile (16 rows x 32 K) of qkh into LDS: one u32 per thread.
  {
    const int rr = t >> 4, pc = (t & 15) * 2;
    const _Float16* src = qkh + (size_t)(tm * 16 + rr) * (2 * DQK) + cofs + pc;
    *(unsigned int*)(Ah + rr * 32 + pc) = *(const unsigned int*)src;
  }
  __syncthreads();

  // 8 waves x 18 column tiles each
  {
    const int wv = t >> 5;
    const v16h a = frag_ld_f16(Ah, 32);
    const int col0 = cd_col();
    const int r0 = cd_row0();
    for (int tn = wv; tn < LQ / 16; tn += 8) {
      v16h bfrag = frag_ld_f16(qkh + (size_t)(tn * 16) * (2 * DQK) + cofs, 2 * DQK);
      v8f acc = {};
      acc = wmma32(a, bfrag, acc);
      const int col = tn * 16 + col0;
#pragma unroll
      for (int i = 0; i < 8; ++i)
        S[(r0 + i) * SROW + col] = acc[i] * INV_TSC;
    }
  }
  __syncthreads();

  // Row softmax over the 16 x LQ strip: 16 threads per row.
  const int row = t >> 4, sub = t & 15;
  float m = -3.0e38f;
  for (int j = sub; j < LQ; j += 16) m = fmaxf(m, S[row * SROW + j]);
  red[t] = m; __syncthreads();
  for (int s = 8; s > 0; s >>= 1) {
    if (sub < s) red[t] = fmaxf(red[t], red[t + s]);
    __syncthreads();
  }
  m = red[row * 16];
  float sum = 0.0f;
  for (int j = sub; j < LQ; j += 16) {
    float e = __expf(S[row * SROW + j] - m);
    S[row * SROW + j] = e;
    sum += e;
  }
  __syncthreads();
  red[t] = sum; __syncthreads();
  for (int s = 8; s > 0; s >>= 1) {
    if (sub < s) red[t] += red[t + s];
    __syncthreads();
  }
  if (sub == 0) rowv[row] = 1.0f / red[row * 16];
  __syncthreads();

  // Normalize + single coalesced store of the strip.
  float* out = attn + (size_t)bh * LQ * LQ + (size_t)(tm * 16) * LQ;
  for (int idx = t; idx < 16 * LQ; idx += 256) {
    const int rr = idx / LQ, cc = idx - rr * LQ;
    out[(size_t)rr * LQ + cc] = S[rr * SROW + cc] * rowv[rr];
  }
}

// ---------------------------------------------------------------------------
// AV: out[q,b,h,d] = sum_k attn[b,h,q,k] * v[k,b,h,d], fused * Ug -> gated f32
// One block per (b,h, 16-row strip); all 8 waves share the attn strip, which
// is staged once to LDS as f16 (single f32->f16 convert per element), then
// each wave computes one of the 8 HID column tiles.
// Dynamic LDS: Ah[16 * AROW] f16 (~74KB)
// ---------------------------------------------------------------------------
__global__ void k_av(const float* __restrict__ attn, const _Float16* __restrict__ VgT,
                     const float* __restrict__ Ug, float* __restrict__ gated) {
  extern __shared__ char smem_raw[];
  _Float16* Ah = (_Float16*)smem_raw;

  const int bh = blockIdx.x / (LQ / 16);
  const int tm = blockIdx.x - bh * (LQ / 16);
  const float* Ap = attn + (size_t)bh * LQ * LQ + (size_t)(tm * 16) * LQ;

  // Stage + convert attn strip to f16 in LDS (coalesced).
  for (int idx = threadIdx.x; idx < 16 * LQ; idx += 256) {
    const int rr = idx / LQ, cc = idx - rr * LQ;
    Ah[rr * AROW + cc] = (_Float16)Ap[(size_t)rr * LQ + cc];
  }
  __syncthreads();

  const int tn = threadIdx.x >> 5;   // 0..7 : HID column tile (uniform per wave)
  const _Float16* Bp = VgT + (size_t)bh * HID * LQ + (size_t)(tn * 16) * LQ;
  v8f acc = {};
  for (int k = 0; k < LQ; k += 32) {
    if (k + 64 < LQ) __builtin_prefetch(Bp + k + 64, 0, 0);
    acc = wmma32(frag_ld_f16(Ah + k, AROW), frag_ld_f16(Bp + k, LQ), acc);
  }

  const int b = bh >> 3, hh = bh & 7;
  const int c  = hh * HID + tn * 16 + cd_col();
  const int q0 = tm * 16 + cd_row0();
#pragma unroll
  for (int i = 0; i < 8; ++i) {
    const int r = (q0 + i) * 2 + b;
    const size_t idx = (size_t)r * CDIM + c;
    gated[idx] = acc[i] * Ug[idx];
  }
}

// ---------------------------------------------------------------------------
// Depthwise 5x5 conv (pad 2) on gated [l,b,c] viewed as [b,c,48,48].
// ---------------------------------------------------------------------------
__global__ void k_dwconv(const float* __restrict__ gated, const float* __restrict__ kern,
                         float* __restrict__ convo) {
  const int tid = blockIdx.x * 256 + threadIdx.x;
  if (tid >= MROWS * CDIM) return;
  const int c = tid & (CDIM - 1);
  const int r = tid >> 10;
  const int b = r & 1, l = r >> 1;
  const int x = l % HW, y = l / HW;
  const float* kw = kern + c * 25;
  float s = 0.0f;
#pragma unroll
  for (int ky = 0; ky < 5; ++ky) {
    const int yy = y + ky - 2;
    if (yy < 0 || yy >= HW) continue;
#pragma unroll
    for (int kx = 0; kx < 5; ++kx) {
      const int xx = x + kx - 2;
      if (xx < 0 || xx >= HW) continue;
      s += gated[(size_t)((yy * HW + xx) * 2 + b) * CDIM + c] * kw[ky * 5 + kx];
    }
  }
  convo[tid] = s;
}

// ---------------------------------------------------------------------------
// Proj: out[4608,512] = convo(f32->f16 on load) @ projT + proj_b
// ---------------------------------------------------------------------------
__global__ void k_proj(const float* __restrict__ A, const _Float16* __restrict__ Bt,
                       const float* __restrict__ bias, float* __restrict__ out) {
  const int wave = (blockIdx.x << 3) + (threadIdx.x >> 5);
  if (wave >= (MROWS / 16) * (HALF / 16)) return;
  const int tm = wave / 32, tn = wave - tm * 32;
  const float*    Ap = A  + (size_t)(tm * 16) * CDIM;
  const _Float16* Bp = Bt + (size_t)(tn * 16) * CDIM;
  v8f acc = {};
  for (int k = 0; k < CDIM; k += 32)
    acc = wmma32(frag_ld_f32(Ap + k, CDIM), frag_ld_f16(Bp + k, CDIM), acc);
  const int n  = tn * 16 + cd_col();
  const int r0 = tm * 16 + cd_row0();
  const float bn = bias[n];
#pragma unroll
  for (int i = 0; i < 8; ++i)
    out[(size_t)(r0 + i) * HALF + n] = acc[i] + bn;
}

// ---------------------------------------------------------------------------
extern "C" void kernel_launch(void* const* d_in, const int* in_sizes, int n_in,
                              void* d_out, int out_size, void* d_ws, size_t ws_size,
                              hipStream_t stream) {
  const float* Q      = (const float*)d_in[0];
  const float* V      = (const float*)d_in[2];
  const float* U      = (const float*)d_in[3];
  const float* w_qk   = (const float*)d_in[4];
  const float* b_qk   = (const float*)d_in[5];
  const float* w_v1   = (const float*)d_in[6];
  const float* b_v1   = (const float*)d_in[7];
  const float* w_v2   = (const float*)d_in[8];
  const float* b_v2   = (const float*)d_in[9];
  const float* w_u1   = (const float*)d_in[10];
  const float* b_u1   = (const float*)d_in[11];
  const float* w_u2   = (const float*)d_in[12];
  const float* b_u2   = (const float*)d_in[13];
  const float* dwk    = (const float*)d_in[14];
  const float* proj_w = (const float*)d_in[15];
  const float* proj_b = (const float*)d_in[16];

  float* out0 = (float*)d_out;                          // [4608, 512]
  float* attn = out0 + (size_t)MROWS * HALF;            // [16, 2304, 2304]

  // Workspace carve-out (256B aligned slices)
  char* wsp = (char*)d_ws;
  auto carve = [&](size_t bytes) -> void* {
    void* p = (void*)wsp;
    wsp += (bytes + 255) & ~(size_t)255;
    return p;
  };
  _Float16* Qh    = (_Float16*)carve((size_t)MROWS * DQK  * 2);
  _Float16* Vh    = (_Float16*)carve((size_t)MROWS * HALF * 2);
  _Float16* Uh    = (_Float16*)carve((size_t)MROWS * HALF * 2);
  _Float16* wqkT  = (_Float16*)carve((size_t)DQK * DQK * 2);
  _Float16* wv1T  = (_Float16*)carve((size_t)HALF * 256 * 2);
  _Float16* wv2T  = (_Float16*)carve((size_t)HALF * 256 * 2);
  _Float16* wu1T  = (_Float16*)carve((size_t)HALF * 256 * 2);
  _Float16* wu2T  = (_Float16*)carve((size_t)HALF * 256 * 2);
  _Float16* projT = (_Float16*)carve((size_t)HALF * CDIM * 2);
  _Float16* qkh   = (_Float16*)carve((size_t)MROWS * DQK * 2);
  _Float16* VgT   = (_Float16*)carve((size_t)BS * NH * HID * LQ * 2);
  float*    Ug    = (float*)carve((size_t)MROWS * CDIM * 4);
  float*    gated = (float*)carve((size_t)MROWS * CDIM * 4);
  float*    convo = (float*)carve((size_t)MROWS * CDIM * 4);

  // 1) precision prep
  k_cvt<<<(MROWS * DQK + 255) / 256, 256, 0, stream>>>(Q, Qh, MROWS * DQK);
  k_cvt<<<(MROWS * HALF + 255) / 256, 256, 0, stream>>>(V, Vh, MROWS * HALF);
  k_cvt<<<(MROWS * HALF + 255) / 256, 256, 0, stream>>>(U, Uh, MROWS * HALF);
  k_cvt_tr<<<(DQK * DQK + 255) / 256, 256, 0, stream>>>(w_qk, wqkT, DQK, DQK);
  k_cvt_tr<<<(256 * HALF + 255) / 256, 256, 0, stream>>>(w_v1, wv1T, 256, HALF);
  k_cvt_tr<<<(256 * HALF + 255) / 256, 256, 0, stream>>>(w_v2, wv2T, 256, HALF);
  k_cvt_tr<<<(256 * HALF + 255) / 256, 256, 0, stream>>>(w_u1, wu1T, 256, HALF);
  k_cvt_tr<<<(256 * HALF + 255) / 256, 256, 0, stream>>>(w_u2, wu2T, 256, HALF);
  k_cvt_tr<<<(CDIM * HALF + 255) / 256, 256, 0, stream>>>(proj_w, projT, CDIM, HALF);

  // 2) qk = Q @ Wqk + b (shared by q and k)
  k_gemm_qk<<<(MROWS / 16) * (DQK / 16) / 8, 256, 0, stream>>>(Qh, wqkT, b_qk, qkh);

  // 3) gated branches -> VgT (f16, transposed) and Ug (f32)
  const int gblk = (MROWS / 16) * (HALF / 16) / 8;      // 1152
  k_gemm_gated_v<<<gblk, 256, 0, stream>>>(Vh,        wv1T, b_v1, VgT, 0);
  k_gemm_gated_v<<<gblk, 256, 0, stream>>>(Vh + 256,  wv2T, b_v2, VgT, 64);
  k_gemm_gated_u<<<gblk, 256, 0, stream>>>(Uh,        wu1T, b_u1, Ug, 0);
  k_gemm_gated_u<<<gblk, 256, 0, stream>>>(Uh + 256,  wu2T, b_u2, Ug, 64);

  // 4) fused scores + softmax -> attn (one write pass over 340MB)
  const size_t sh_sc = (size_t)16 * SROW * 4 + (size_t)16 * 32 * 2
                     + (size_t)256 * 4 + (size_t)16 * 4;      // ~150KB
  k_scores_softmax<<<16 * (LQ / 16), 256, sh_sc, stream>>>(qkh, attn);

  // 5) attn @ v (LDS-staged attn strip), fused * Ug
  const size_t sh_av = (size_t)16 * AROW * 2;                 // ~74KB
  k_av<<<16 * (LQ / 16), 256, sh_av, stream>>>(attn, VgT, Ug, gated);

  // 6) depthwise 5x5 conv
  k_dwconv<<<(MROWS * CDIM + 255) / 256, 256, 0, stream>>>(gated, dwk, convo);

  // 7) final projection -> d_out[0 : 4608*512]
  k_proj<<<(MROWS / 16) * (HALF / 16) / 8, 256, 0, stream>>>(convo, projT, proj_b, out0);
}
